// MultiHeadGraphAttentionLayer_60662118088932
// MI455X (gfx1250) — compile-verified, matrix operand order checked by reading, not run on hardware
//
#include <hip/hip_runtime.h>

#define B_  8
#define N_  1024
#define DIN 256
#define DOUT 64
#define H_  8
#define NEG_SLOPE 0.2f

typedef __attribute__((ext_vector_type(16))) __bf16 v16bf;
typedef __attribute__((ext_vector_type(2)))  __bf16 bf16x2;
typedef __attribute__((ext_vector_type(8)))  float  v8f;

// K index for VGPR-pair i (0..7) and lane half (0/1), per CDNA5 16-bit A/B layout:
// i=0..3 -> K={0,2,4,6}, i=4..7 -> K={16,18,20,22}, +8 for lanes 16..31.
__device__ __forceinline__ int kmap(int i, int half) {
    return ((i & 3) << 1) + ((i >> 2) << 4) + (half << 3);
}

__device__ __forceinline__ __bf16 f2bf(float f) {
    unsigned int u = __builtin_bit_cast(unsigned int, f);
    unsigned int r = (u + 0x7FFFu + ((u >> 16) & 1u)) >> 16;  // RNE
    return __builtin_bit_cast(__bf16, (unsigned short)r);
}

__device__ __forceinline__ unsigned lds_off(const void* p) {
    // generic pointers to LDS carry the offset in the low 32 bits (ISA 10.2 aperture map)
    return (unsigned)(uintptr_t)p;
}

// ---------------- kernel 0a: W (H,DIN,DOUT) f32 -> Wt (H,DOUT,DIN) bf16 ----------------
__global__ void gat_prep_w_kernel(const float* __restrict__ W, __bf16* __restrict__ Wt) {
    int e = blockIdx.x * 256 + threadIdx.x;          // 0 .. H*DOUT*DIN-1
    int d = e & (DIN - 1);
    int f = (e >> 8) & (DOUT - 1);
    int hh = e >> 14;
    Wt[e] = f2bf(W[((size_t)hh * DIN + d) * DOUT + f]);
}

// ---------------- kernel 0b: h f32 -> bf16 (same layout) ----------------
__global__ void gat_prep_h_kernel(const float* __restrict__ h, __bf16* __restrict__ hB) {
    int e = blockIdx.x * 256 + threadIdx.x;          // 0 .. B*N*DIN-1
    hB[e] = f2bf(h[e]);
}

// ---------------- kernel 1: Wh = h@W via WMMA; emit WhT_bf16, f1, f2 ----------------
// grid: B*H*(N/128) blocks of 256 threads (8 waves); wave w owns rows [tile*128+w*16, +16)
__global__ void gat_wh_kernel(const __bf16* __restrict__ hB,
                              const __bf16* __restrict__ Wt,
                              const float* __restrict__ a,
                              __bf16* __restrict__ WhT,
                              float* __restrict__ f1,
                              float* __restrict__ f2) {
    __shared__ float lds[8][16][65];  // padded to dodge bank conflicts on column reads

    const int tid  = threadIdx.x;
    const int wave = tid >> 5;
    const int lane = tid & 31;
    const int r    = lane & 15;
    const int half = lane >> 4;

    const int blk  = blockIdx.x;         // b*64 + hh*8 + tile
    const int tile = blk & 7;
    const int hh   = (blk >> 3) & 7;
    const int b    = blk >> 6;
    const int n0   = tile * 128 + wave * 16;
    const int bh   = b * H_ + hh;

    v8f acc[4];
    for (int t = 0; t < 4; ++t)
        for (int v = 0; v < 8; ++v) acc[t][v] = 0.f;

    const __bf16* hrow = hB + ((size_t)(b * N_ + n0 + r)) * DIN;
    for (int kd = 0; kd < DIN; kd += 32) {
        v16bf afrag;
        for (int i = 0; i < 8; ++i) {
            bf16x2 hv = *(const bf16x2*)(hrow + kd + kmap(i, half));
            afrag[2 * i]     = hv.x;
            afrag[2 * i + 1] = hv.y;
        }
        for (int t = 0; t < 4; ++t) {
            v16bf bfrag;
            const __bf16* wp = Wt + ((size_t)(hh * DOUT + t * 16 + r)) * DIN + kd;
            for (int i = 0; i < 8; ++i) {
                bf16x2 wv = *(const bf16x2*)(wp + kmap(i, half));
                bfrag[2 * i]     = wv.x;
                bfrag[2 * i + 1] = wv.y;
            }
            acc[t] = __builtin_amdgcn_wmma_f32_16x16x32_bf16(
                false, afrag, false, bfrag, (short)0, acc[t], false, false);
        }
    }

    // spill 16x64 tile to LDS (C/D layout: row = v + 8*half, col = 16t + r)
    for (int t = 0; t < 4; ++t)
        for (int v = 0; v < 8; ++v)
            lds[wave][v + 8 * half][t * 16 + r] = acc[t][v];
    __syncthreads();

    // lanes 0-15: f1 of row r; lanes 16-31: f2 of row r
    {
        const float* av = a + hh * (2 * DOUT) + half * DOUT;
        float s = 0.f;
        for (int f = 0; f < DOUT; ++f) s += lds[wave][r][f] * av[f];
        float* fout = half ? f2 : f1;
        fout[(size_t)bh * N_ + n0 + r] = s;
    }

    // WhT (B,H,DOUT,N) bf16, packed (n,n+1) pairs for contiguous K loads in GEMM2
    for (int i = 0; i < 16; ++i) {
        int idx = i * 32 + lane;       // 0..511 : f = idx>>3, pair j = idx&7
        int f   = idx >> 3;
        int nl  = (idx & 7) * 2;
        bf16x2 p;
        p.x = f2bf(lds[wave][nl][f]);
        p.y = f2bf(lds[wave][nl + 1][f]);
        *(bf16x2*)(WhT + ((size_t)(bh * DOUT + f)) * N_ + n0 + nl) = p;
    }
}

// ---------------- kernel 2: fused masked-softmax attention (flash style) ----------------
// grid: B*(N/32) blocks of 256 threads; wave w = head w; each wave owns 32 query rows
// (two 16-row A sets sharing the B fragments). adj/f2 tiles land in LDS via async loads.
__global__ void gat_attn_kernel(const int* __restrict__ adj,
                                const __bf16* __restrict__ WhT,
                                const float* __restrict__ f1,
                                const float* __restrict__ f2,
                                float* __restrict__ out) {
    __shared__ int   adjT[32][32];
    __shared__ float f2T[8][32];

    const int tid  = threadIdx.x;
    const int hh   = tid >> 5;          // wave == head
    const int lane = tid & 31;
    const int r    = lane & 15;
    const int half = lane >> 4;

    const int blk  = blockIdx.x;        // b*32 + tile
    const int tile = blk & 31;
    const int b    = blk >> 5;
    const int n0   = tile * 32;
    const int bh   = b * H_ + hh;

    float f1r[2], m_run[2], l_run[2];
    v8f acc[2][4];
    for (int s = 0; s < 2; ++s) {
        f1r[s]  = f1[(size_t)bh * N_ + n0 + s * 16 + r];
        m_run[s] = -1e30f;
        l_run[s] = 0.f;
        for (int t = 0; t < 4; ++t)
            for (int v = 0; v < 8; ++v) acc[s][t][v] = 0.f;
    }

    const int* adjbase = adj + (size_t)b * N_ * N_;
    const int  arow    = tid >> 3;          // 0..31
    const int  acol    = (tid & 7) * 4;     // 0,4,..,28

    for (int m0 = 0; m0 < N_; m0 += 32) {
        __syncthreads();   // previous iteration done reading LDS
        {   // async tile fill: adj 32x32 ints (b128/thread), f2 8x32 floats (b32/thread)
            unsigned long long ga =
                (unsigned long long)(adjbase + (size_t)(n0 + arow) * N_ + m0 + acol);
            unsigned la = lds_off(&adjT[arow][acol]);
            asm volatile("global_load_async_to_lds_b128 %0, %1, off"
                         :: "v"(la), "v"(ga) : "memory");
            unsigned long long gf =
                (unsigned long long)(f2 + (size_t)(b * H_ + (tid >> 5)) * N_ + m0 + (tid & 31));
            unsigned lf = lds_off(&f2T[tid >> 5][tid & 31]);
            asm volatile("global_load_async_to_lds_b32 %0, %1, off"
                         :: "v"(lf), "v"(gf) : "memory");
            asm volatile("s_wait_asynccnt 0" ::: "memory");
        }
        __syncthreads();

        if (m0 + 32 < N_)  // gfx1250 global_prefetch_b8 on next adj tile
            __builtin_prefetch(adjbase + (size_t)(n0 + arow) * N_ + m0 + 32, 0, 1);

        v16bf pfrag[2];
        for (int s = 0; s < 2; ++s) {
            // scores e = leakyrelu(f1[n]+f2[m]) masked; row = s*16+r, K order = A-frag order
            float ev[16];
            float mx = -1e30f;
            for (int i = 0; i < 8; ++i) {
                int k = kmap(i, half);
                float x0 = f1r[s] + f2T[hh][k];
                float x1 = f1r[s] + f2T[hh][k + 1];
                x0 = x0 >= 0.f ? x0 : NEG_SLOPE * x0;
                x1 = x1 >= 0.f ? x1 : NEG_SLOPE * x1;
                ev[2 * i]     = adjT[s * 16 + r][k]     ? x0 : -1e30f;
                ev[2 * i + 1] = adjT[s * 16 + r][k + 1] ? x1 : -1e30f;
                mx = fmaxf(mx, fmaxf(ev[2 * i], ev[2 * i + 1]));
            }
            mx = fmaxf(mx, __shfl_xor(mx, 16));       // combine halves -> row max
            float m_new = fmaxf(m_run[s], mx);
            float scale = __expf(m_run[s] - m_new);

            // rescale accumulators: acc row = v + 8*half, scale lives at lane (row&15)
            for (int v = 0; v < 8; ++v) {
                float sR = __shfl(scale, v + 8 * half);
                for (int t = 0; t < 4; ++t) acc[s][t][v] *= sR;
            }

            float psum = 0.f;
            for (int j = 0; j < 16; ++j) {
                float p = __expf(ev[j] - m_new);
                psum += p;
                pfrag[s][j] = f2bf(p);
            }
            psum += __shfl_xor(psum, 16);
            l_run[s] = l_run[s] * scale + psum;
            m_run[s] = m_new;
        }

        // acc[s] += p[s] (16x32) @ WhT-slice (32x64): 4 B-frags shared by both row sets
        const __bf16* wb = WhT + (size_t)bh * DOUT * N_ + m0;
        for (int t = 0; t < 4; ++t) {
            v16bf bfrag;
            const __bf16* wp = wb + (size_t)(t * 16 + r) * N_;
            for (int i = 0; i < 8; ++i) {
                bf16x2 wv = *(const bf16x2*)(wp + kmap(i, half));
                bfrag[2 * i]     = wv.x;
                bfrag[2 * i + 1] = wv.y;
            }
            acc[0][t] = __builtin_amdgcn_wmma_f32_16x16x32_bf16(
                false, pfrag[0], false, bfrag, (short)0, acc[0][t], false, false);
            acc[1][t] = __builtin_amdgcn_wmma_f32_16x16x32_bf16(
                false, pfrag[1], false, bfrag, (short)0, acc[1][t], false, false);
        }
    }

    // normalize and store: out (B, N, H*DOUT)
    for (int s = 0; s < 2; ++s) {
        float inv = 1.f / l_run[s];   // for row s*16+r (duplicated across halves)
        for (int v = 0; v < 8; ++v) {
            float iR = __shfl(inv, v + 8 * half);
            int   n  = n0 + s * 16 + v + 8 * half;
            float* op = out + ((size_t)(b * N_ + n)) * (H_ * DOUT) + hh * DOUT;
            for (int t = 0; t < 4; ++t)
                op[t * 16 + r] = acc[s][t][v] * iR;
        }
    }
}

extern "C" void kernel_launch(void* const* d_in, const int* in_sizes, int n_in,
                              void* d_out, int out_size, void* d_ws, size_t ws_size,
                              hipStream_t stream) {
    const float* h   = (const float*)d_in[0];
    const int*   adj = (const int*)d_in[1];
    const float* W   = (const float*)d_in[2];
    const float* a   = (const float*)d_in[3];
    float* out = (float*)d_out;

    char* ws = (char*)d_ws;
    const size_t wt_bytes  = (size_t)H_ * DOUT * DIN * 2;        // 256 KB
    const size_t hb_bytes  = (size_t)B_ * N_ * DIN * 2;          // 4 MB
    const size_t wht_bytes = (size_t)B_ * H_ * DOUT * N_ * 2;    // 8 MB
    const size_t f_bytes   = (size_t)B_ * H_ * N_ * 4;           // 256 KB
    __bf16* Wt  = (__bf16*)ws;
    __bf16* hBf = (__bf16*)(ws + wt_bytes);
    __bf16* WhT = (__bf16*)(ws + wt_bytes + hb_bytes);
    float*  f1  = (float*)(ws + wt_bytes + hb_bytes + wht_bytes);
    float*  f2  = (float*)(ws + wt_bytes + hb_bytes + wht_bytes + f_bytes);

    gat_prep_w_kernel<<<(H_ * DOUT * DIN) / 256, 256, 0, stream>>>(W, Wt);
    gat_prep_h_kernel<<<(B_ * N_ * DIN) / 256, 256, 0, stream>>>(h, hBf);
    gat_wh_kernel<<<B_ * H_ * (N_ / 128), 256, 0, stream>>>(hBf, Wt, a, WhT, f1, f2);
    gat_attn_kernel<<<B_ * (N_ / 32), 256, 0, stream>>>(adj, WhT, f1, f2, out);
}
